// BoundedLinearAttention_85950885528257
// MI455X (gfx1250) — matrix operand order repaired and checked by reading, not compiled, not run on hardware
//
#include <hip/hip_runtime.h>
#include <cstddef>
#include <cstdint>

typedef _Float16 half_t;
typedef __attribute__((ext_vector_type(16))) _Float16 v16h;
typedef __attribute__((ext_vector_type(8)))  _Float16 v8h;
typedef __attribute__((ext_vector_type(8)))  float    v8f;

union V16 { v16h v; v8h h[2]; };

#define B_  4
#define N_  4096
#define C_  1024
#define H_  16
#define D_  64
#define BH_ (B_ * H_)     // 64
#define M_  (B_ * N_)     // 16384

#define TMM 256           // block tile rows (M)
#define TNN 128           // block tile cols (N)
#define TK  32            // K step per WMMA
#define LDA 40            // LDS row stride in halves (80B: 16B-aligned, bank-spread)
#define KVSPLIT 8         // N-splits for kv reduction

// ---------------------------------------------------------------------------
// fp32 -> f16 conversion (grid-stride)
// ---------------------------------------------------------------------------
__global__ __launch_bounds__(256) void cvt_f16_kernel(const float* __restrict__ in,
                                                      half_t* __restrict__ out, int n) {
  int i = blockIdx.x * blockDim.x + threadIdx.x;
  int stride = gridDim.x * blockDim.x;
  for (; i < n; i += stride) out[i] = (half_t)in[i];
}

// ---------------------------------------------------------------------------
// Generic WMMA GEMM:  Y(M x Nout) = A16(M x K) @ B16(Nout x K)^T
//   EPI == 0 : QKV epilogue -> split/scale into q16/k16/v16, head-major (B,H,N,D)
//   EPI == 1 : final epilogue -> fp32 out + bias
// Block: 256 threads (8 waves). Tile 256x128. Wave: 64(M) x 64(N) = 4x4 WMMA tiles.
// Tiles staged with GLOBAL_LOAD_ASYNC_TO_LDS_B128 (ASYNCcnt), double-buffered.
// ---------------------------------------------------------------------------
template <int EPI>
__global__ __launch_bounds__(256) void gemm_kernel(const half_t* __restrict__ Ag,
                                                   const half_t* __restrict__ Bg,
                                                   int K,
                                                   const float* __restrict__ aux, // fs (EPI=0) or bias (EPI=1)
                                                   half_t* __restrict__ o16a,
                                                   half_t* __restrict__ o16b,
                                                   half_t* __restrict__ o16c,
                                                   float* __restrict__ of32) {
  __shared__ __align__(16) half_t As[2][TMM * LDA];   // 2 x 20 KB
  __shared__ __align__(16) half_t Bs[2][TNN * LDA];   // 2 x 10 KB

  const int rowBase = blockIdx.x * TMM;   // over M
  const int colBase = blockIdx.y * TNN;   // over Nout
  const int tid  = threadIdx.x;
  const int wave = tid >> 5, lane = tid & 31;
  const int wm = wave & 3, wn = wave >> 2;      // 4x2 wave grid -> 64x64 each
  const int hs = lane >> 4, lr = lane & 15;     // half-wave select, row-in-tile

  // async global -> LDS copy of one 256x32 A tile and 128x32 B tile (16B chunks)
  auto async_tiles = [&](int k0, int buf) {
#pragma unroll
    for (int i = 0; i < 4; i++) {               // A: 1024 chunks / 256 threads
      int chunk = i * 256 + tid;
      int row = chunk >> 2;
      int kc  = (chunk & 3) * 8;
      unsigned lA = (unsigned)(size_t)&As[buf][row * LDA + kc];
      const half_t* gA = &Ag[(size_t)(rowBase + row) * K + k0 + kc];
      asm volatile("global_load_async_to_lds_b128 %0, %1, off"
                   :: "v"(lA), "v"(gA) : "memory");
    }
#pragma unroll
    for (int i = 0; i < 2; i++) {               // B: 512 chunks / 256 threads
      int chunk = i * 256 + tid;
      int row = chunk >> 2;
      int kc  = (chunk & 3) * 8;
      unsigned lB = (unsigned)(size_t)&Bs[buf][row * LDA + kc];
      const half_t* gB = &Bg[(size_t)(colBase + row) * K + k0 + kc];
      asm volatile("global_load_async_to_lds_b128 %0, %1, off"
                   :: "v"(lB), "v"(gB) : "memory");
    }
  };

  v8f acc[4][4] = {};

  async_tiles(0, 0);
  asm volatile("s_wait_asynccnt 0x0" ::: "memory");
  __syncthreads();

  const int nk = K / TK;
  for (int kt = 0; kt < nk; kt++) {
    const int buf = kt & 1;
    if (kt + 1 < nk) async_tiles((kt + 1) * TK, buf ^ 1);  // prefetch in flight

    V16 b[4];
#pragma unroll
    for (int j = 0; j < 4; j++) {
      // B 32x16 f16 layout: lane lr holds column, chunks at k = hs*8 and hs*8+16
      const half_t* p = &Bs[buf][(wn * 64 + j * 16 + lr) * LDA + hs * 8];
      b[j].h[0] = *(const v8h*)p;
      b[j].h[1] = *(const v8h*)(p + 16);
    }
#pragma unroll
    for (int i = 0; i < 4; i++) {
      // A 16x32 f16 layout: lane lr holds row, chunks at k = hs*8 and hs*8+16
      V16 a;
      const half_t* p = &As[buf][(wm * 64 + i * 16 + lr) * LDA + hs * 8];
      a.h[0] = *(const v8h*)p;
      a.h[1] = *(const v8h*)(p + 16);
#pragma unroll
      for (int j = 0; j < 4; j++)
        acc[i][j] = __builtin_amdgcn_wmma_f32_16x16x32_f16(
            false, a.v, false, b[j].v, (short)0, acc[i][j], false, false);
    }
    asm volatile("s_wait_asynccnt 0x0" ::: "memory");  // prefetch landed
    __syncthreads();                                    // visible to all waves
  }

  // Epilogue. C/D layout: VGPR e -> row e + 8*hs, lane lr -> column.
#pragma unroll
  for (int i = 0; i < 4; i++)
#pragma unroll
    for (int j = 0; j < 4; j++)
#pragma unroll
      for (int e = 0; e < 8; e++) {
        int r = rowBase + wm * 64 + i * 16 + e + 8 * hs;
        int c = colBase + wn * 64 + j * 16 + lr;
        float val = acc[i][j][e];
        if (EPI == 0) {
          int bb = r >> 12, n = r & (N_ - 1);
          int s  = c >> 10, rem = c & (C_ - 1);
          int h  = rem >> 6, d = rem & (D_ - 1);
          size_t idx = (((size_t)bb * H_ + h) * N_ + n) * D_ + d;
          if (s == 0)      o16a[idx] = (half_t)(val * aux[d] * 0.125f); // q * fs / sqrt(64)
          else if (s == 1) o16b[idx] = (half_t)val;                     // k (fs applied later)
          else             o16c[idx] = (half_t)val;                     // v
        } else {
          of32[(size_t)r * C_ + c] = val + aux[c];
        }
      }
}

// ---------------------------------------------------------------------------
// kv_part: block (bh, split) computes partial kv[d][f] and k_sum[d] over 512 rows.
// Deterministic two-stage reduction (no float atomics -> graph-replay stable).
// ---------------------------------------------------------------------------
__global__ __launch_bounds__(256) void kv_part_kernel(const half_t* __restrict__ k16,
                                                      const half_t* __restrict__ v16,
                                                      const unsigned char* __restrict__ mask,
                                                      const float* __restrict__ fs,
                                                      float* __restrict__ part_kv,
                                                      float* __restrict__ part_ks) {
  const int bh = blockIdx.x;          // 0..63
  const int sp = blockIdx.y;          // 0..KVSPLIT-1
  const int b  = bh >> 4;             // H_=16
  __shared__ float kt[64][65];
  __shared__ float vt[64][65];

  const int tid = threadIdx.x;
  const int d = tid & 63, g = tid >> 6;   // thread owns (d, f-group g)
  float accv[16];
#pragma unroll
  for (int j = 0; j < 16; j++) accv[j] = 0.f;
  float asum = 0.f;

  const size_t base = (size_t)bh * N_ * D_;
  const int rows = N_ / KVSPLIT;          // 512
  for (int c = 0; c < rows / 64; c++) {
    const int n0 = sp * rows + c * 64;
#pragma unroll
    for (int i = 0; i < 16; i++) {
      int e = i * 256 + tid;
      int row = e >> 6, col = e & 63;
      float m = mask[b * N_ + n0 + row] ? 1.f : 0.f;
      kt[row][col] = m * fs[col] * (float)k16[base + (size_t)(n0 + row) * D_ + col];
      vt[row][col] = m * (float)v16[base + (size_t)(n0 + row) * D_ + col];
    }
    __syncthreads();
    for (int n = 0; n < 64; n++) {
      float kk = kt[n][d];
      if (g == 0) asum += kk;
#pragma unroll
      for (int j = 0; j < 16; j++) accv[j] += kk * vt[n][g * 16 + j];
    }
    __syncthreads();
  }

#pragma unroll
  for (int j = 0; j < 16; j++)
    part_kv[((size_t)(bh * KVSPLIT + sp)) * (D_ * D_) + (size_t)d * D_ + g * 16 + j] = accv[j];
  if (g == 0) part_ks[(bh * KVSPLIT + sp) * D_ + d] = asum;
}

// ---------------------------------------------------------------------------
// kv_final: reduce partials, store kv^T (f-major, f16) and 1/(k_sum+1e-8).
// ---------------------------------------------------------------------------
__global__ __launch_bounds__(256) void kv_final_kernel(const float* __restrict__ part_kv,
                                                       const float* __restrict__ part_ks,
                                                       half_t* __restrict__ kvT16,
                                                       float* __restrict__ rks) {
  const int bh = blockIdx.x;
  const int tid = threadIdx.x;
#pragma unroll
  for (int t = 0; t < 16; t++) {
    int e = t * 256 + tid;                  // kv element: d*64 + f
    float s = 0.f;
#pragma unroll
    for (int p = 0; p < KVSPLIT; p++)
      s += part_kv[((size_t)(bh * KVSPLIT + p)) * (D_ * D_) + e];
    int d = e >> 6, f = e & 63;
    kvT16[(size_t)bh * (D_ * D_) + (size_t)f * D_ + d] = (half_t)s;   // transposed
  }
  if (tid < D_) {
    float s = 0.f;
#pragma unroll
    for (int p = 0; p < KVSPLIT; p++)
      s += part_ks[(bh * KVSPLIT + p) * D_ + tid];
    rks[bh * D_ + tid] = 1.f / (s + 1e-8f);
  }
}

// ---------------------------------------------------------------------------
// attn_out: out[b,h,n,f] = (q[b,h,n,:] @ kv[:,f]) * rks[f]  via WMMA.
// Block = 128 rows of q for one (b,h); kv^T (64x64 f16) cached in LDS.
// ---------------------------------------------------------------------------
__global__ __launch_bounds__(256) void attn_out_kernel(const half_t* __restrict__ q16,
                                                       const half_t* __restrict__ kvT16,
                                                       const float* __restrict__ rks,
                                                       half_t* __restrict__ op16) {
  const int n0 = blockIdx.x * 128;
  const int bh = blockIdx.y;
  __shared__ __align__(16) half_t kvs[64 * 80];  // stride 80 halves = 160B (16B-aligned)
  __shared__ float rk[64];

  const int tid = threadIdx.x;
#pragma unroll
  for (int i = 0; i < 2; i++) {
    int chunk = tid * 2 + i;            // 0..511 x 8 halves
    int row = chunk >> 3, col = (chunk & 7) * 8;
    *(v8h*)&kvs[row * 80 + col] =
        *(const v8h*)&kvT16[(size_t)bh * (D_ * D_) + (size_t)row * D_ + col];
  }
  if (tid < 64) rk[tid] = rks[bh * D_ + tid];
  __syncthreads();

  const int wave = tid >> 5, lane = tid & 31;
  const int hs = lane >> 4, lr = lane & 15;

  v8f acc[4] = {};
  const half_t* qb = q16 + ((size_t)bh * N_ + n0 + wave * 16) * D_;
#pragma unroll
  for (int ks = 0; ks < 2; ks++) {
    V16 a;
    const half_t* p = &qb[(size_t)lr * D_ + ks * 32 + hs * 8];
    a.h[0] = *(const v8h*)p;
    a.h[1] = *(const v8h*)(p + 16);
#pragma unroll
    for (int j = 0; j < 4; j++) {
      V16 bf;
      const half_t* q2 = &kvs[(j * 16 + lr) * 80 + ks * 32 + hs * 8];
      bf.h[0] = *(const v8h*)q2;
      bf.h[1] = *(const v8h*)(q2 + 16);
      acc[j] = __builtin_amdgcn_wmma_f32_16x16x32_f16(
          false, a.v, false, bf.v, (short)0, acc[j], false, false);
    }
  }

  const int b = bh >> 4, h = bh & 15;
#pragma unroll
  for (int j = 0; j < 4; j++)
#pragma unroll
    for (int e = 0; e < 8; e++) {
      int n = n0 + wave * 16 + e + 8 * hs;
      int f = j * 16 + lr;
      float val = acc[j][e] * rk[f];
      op16[((size_t)(b * N_ + n)) * C_ + h * D_ + f] = (half_t)val;
    }
}

// ---------------------------------------------------------------------------
extern "C" void kernel_launch(void* const* d_in, const int* in_sizes, int n_in,
                              void* d_out, int out_size, void* d_ws, size_t ws_size,
                              hipStream_t stream) {
  const float* x            = (const float*)d_in[0];
  const unsigned char* mask = (const unsigned char*)d_in[1];
  const float* qkv_w        = (const float*)d_in[2];
  const float* out_w        = (const float*)d_in[3];
  const float* out_b        = (const float*)d_in[4];
  const float* fs           = (const float*)d_in[5];
  float* out = (float*)d_out;

  char* ws = (char*)d_ws;
  size_t off = 0;
  auto alloc = [&](size_t bytes) -> void* {
    void* p = ws + off;
    off += (bytes + 255) & ~(size_t)255;
    return p;
  };
  half_t* x16   = (half_t*)alloc((size_t)M_ * C_ * 2);
  half_t* qw16  = (half_t*)alloc((size_t)3 * C_ * C_ * 2);
  half_t* ow16  = (half_t*)alloc((size_t)C_ * C_ * 2);
  half_t* q16   = (half_t*)alloc((size_t)BH_ * N_ * D_ * 2);
  half_t* k16   = (half_t*)alloc((size_t)BH_ * N_ * D_ * 2);
  half_t* v16   = (half_t*)alloc((size_t)BH_ * N_ * D_ * 2);
  half_t* kvT16 = (half_t*)alloc((size_t)BH_ * D_ * D_ * 2);
  float*  rks   = (float*)alloc((size_t)BH_ * D_ * 4);
  half_t* op16  = (half_t*)alloc((size_t)M_ * C_ * 2);
  float*  pkv   = (float*)alloc((size_t)BH_ * KVSPLIT * D_ * D_ * 4);
  float*  pks   = (float*)alloc((size_t)BH_ * KVSPLIT * D_ * 4);

  cvt_f16_kernel<<<2048, 256, 0, stream>>>(x, x16, M_ * C_);
  cvt_f16_kernel<<<1024, 256, 0, stream>>>(qkv_w, qw16, 3 * C_ * C_);
  cvt_f16_kernel<<<512, 256, 0, stream>>>(out_w, ow16, C_ * C_);

  gemm_kernel<0><<<dim3(M_ / TMM, 3 * C_ / TNN), 256, 0, stream>>>(
      x16, qw16, C_, fs, q16, k16, v16, nullptr);

  kv_part_kernel<<<dim3(BH_, KVSPLIT), 256, 0, stream>>>(k16, v16, mask, fs, pkv, pks);
  kv_final_kernel<<<BH_, 256, 0, stream>>>(pkv, pks, kvT16, rks);

  attn_out_kernel<<<dim3(N_ / 128, BH_), 256, 0, stream>>>(q16, kvT16, rks, op16);

  gemm_kernel<1><<<dim3(M_ / TMM, C_ / TNN), 256, 0, stream>>>(
      op16, ow16, C_, out_b, nullptr, nullptr, nullptr, out);
}
